// EvoformerBlockV47_42374147342905
// MI455X (gfx1250) — matrix operand order repaired and checked by reading, not compiled, not run on hardware
//
#include <hip/hip_runtime.h>
#include <math.h>

typedef __attribute__((ext_vector_type(16))) __bf16 v16bf;
typedef __attribute__((ext_vector_type(8)))  __bf16 v8bf;
typedef __attribute__((ext_vector_type(4)))  __bf16 bf16x4;
typedef __attribute__((ext_vector_type(2)))  __bf16 bf16x2;
typedef __attribute__((ext_vector_type(8)))  float  v8f;

// ---------------- problem dims ----------------
constexpr int Sdim = 64, Ndim = 256, Cdim = 256, Pdim = 128, Hn = 8, HDm = 32, RK = 32;
constexpr int SN = Sdim * Ndim;        // 16384
constexpr int NNi = Ndim * Ndim;       // 65536
constexpr long SZ_HEAD = (long)SN * Cdim;         // 4194304
constexpr long SZ_QKV  = (long)SN * 3 * Cdim;     // 12582912
constexpr long SZ_SC   = (long)Sdim * Hn * Ndim * Ndim; // 33554432
constexpr long SZ_NNR  = (long)NNi * RK;          // 2097152

// ---------------- input index map (setup_inputs dict order) ----------------
enum {
  IN_MSA = 0, IN_PAIR, IN_MASK,
  P_ROW_QKV_W, P_ROW_QKV_B, P_ROW_OUT_W, P_ROW_OUT_B, P_ROW_LN_G, P_ROW_LN_B,
  P_ROW_PB_W, P_ROW_PB_B,
  P_COL_QKV_W, P_COL_QKV_B, P_COL_OUT_W, P_COL_OUT_B, P_COL_LN_G, P_COL_LN_B,
  P_OPM_W, P_OPM_B, P_OPM_LN_G, P_OPM_LN_B,
  P_TO_L_W, P_TO_L_B, P_TO_R_W, P_TO_R_B, P_TO_G_W, P_TO_G_B, P_TO_E_W, P_TO_E_B,
  P_TO_O_W, P_TO_O_B, P_TO_N_G, P_TO_N_B,
  P_TI_L_W, P_TI_L_B, P_TI_R_W, P_TI_R_B, P_TI_G_W, P_TI_G_B,
  P_TI_O_W, P_TI_O_B, P_TI_N_G, P_TI_N_B,
  P_T1_W, P_T1_B, P_T2_W, P_T2_B, P_TN_G, P_TN_B, P_PF_G, P_PF_B
};

// ---------------- reductions ----------------
__device__ __forceinline__ float wave_red_sum(float v) {
#pragma unroll
  for (int off = 16; off > 0; off >>= 1) v += __shfl_xor(v, off, 32);
  return v;
}
__device__ __forceinline__ float wave_red_max(float v) {
#pragma unroll
  for (int off = 16; off > 0; off >>= 1) v = fmaxf(v, __shfl_xor(v, off, 32));
  return v;
}
__device__ __forceinline__ float block_red_sum(float v, float* red) {
  int lane = threadIdx.x & 31, w = threadIdx.x >> 5;
  int nw = (blockDim.x + 31) >> 5;
  v = wave_red_sum(v);
  if (lane == 0) red[w] = v;
  __syncthreads();
  if (threadIdx.x == 0) {
    float r = 0.f;
    for (int i = 0; i < nw; ++i) r += red[i];
    red[8] = r;
  }
  __syncthreads();
  return red[8];
}
__device__ __forceinline__ float block_red_max(float v, float* red) {
  int lane = threadIdx.x & 31, w = threadIdx.x >> 5;
  int nw = (blockDim.x + 31) >> 5;
  v = wave_red_max(v);
  if (lane == 0) red[w] = v;
  __syncthreads();
  if (threadIdx.x == 0) {
    float r = -3.4e38f;
    for (int i = 0; i < nw; ++i) r = fmaxf(r, red[i]);
    red[8] = r;
  }
  __syncthreads();
  return red[8];
}

// ---------------- WMMA GEMM ----------------
// D[M,Nc] = epilogue( A[M,K] @ B[K,Nc] ), bf16 inputs, f32 accumulate.
// TRANSB=false: B stored row-major (K x Nc).  TRANSB=true: B stored (Nc x K).
enum { EPI_BIAS = 0, EPI_BIAS_RELU = 1, EPI_BIAS_SIG = 2, EPI_SCALE = 3, EPI_NONE = 4 };

constexpr int BM = 64, BN = 64, BK = 32;
constexpr int LP = BK + 8;   // LDS pitch: 40 bf16 = 80 B (multiple of 16 B)

template <bool TRANSB, int EPI>
__global__ __launch_bounds__(128) void gemm_wmma(
    const float* __restrict__ A, const float* __restrict__ B,
    const float* __restrict__ bias, float* __restrict__ D,
    int M, int Nc, int K, int lda, int ldb, int ldd,
    long aBatch, long bBatch, long dBatch, float scale) {
  __shared__ __attribute__((aligned(16))) __bf16 As[BM][LP];
  __shared__ __attribute__((aligned(16))) __bf16 Bs[BN][LP];

  const int batch = blockIdx.z;
  const float* Ab = A + (long)batch * aBatch;
  const float* Bb = B + (long)batch * bBatch;
  float* Db = D + (long)batch * dBatch;

  const int m0 = blockIdx.x * BM, n0 = blockIdx.y * BN;
  const int t = threadIdx.x;
  const int wid = t >> 5, lane = t & 31;
  const int half = lane >> 4, r = lane & 15;

  const bool fullM = (m0 + BM <= M);
  const bool fullN = (n0 + BN <= Nc);

  const v8f vzero = {0.f, 0.f, 0.f, 0.f, 0.f, 0.f, 0.f, 0.f};
  v8f acc[4] = {vzero, vzero, vzero, vzero};

  for (int k0 = 0; k0 < K; k0 += BK) {
    const bool fullK = (k0 + BK <= K);

    // ---------- stage A tile (fp32 -> bf16) ----------
    if (fullM && fullK) {
      // 64x32 floats = 512 float4 chunks; 128 threads x 4 iters.
#pragma unroll
      for (int e = 0; e < 512; e += 128) {
        int idx = e + t;
        int i = idx >> 3;            // row, 8 chunks/row
        int j = (idx & 7) << 2;      // col 0,4,...,28
        float4 f = *(const float4*)&Ab[(long)(m0 + i) * lda + (k0 + j)];
        bf16x4 h;
        h[0] = (__bf16)f.x; h[1] = (__bf16)f.y; h[2] = (__bf16)f.z; h[3] = (__bf16)f.w;
        *(bf16x4*)&As[i][j] = h;
      }
    } else {
#pragma unroll
      for (int e = 0; e < BM * BK; e += 128) {
        int idx = e + t;
        int i = idx >> 5, j = idx & 31;
        float v = 0.f;
        int gm = m0 + i, gk = k0 + j;
        if (gm < M && gk < K) v = Ab[(long)gm * lda + gk];
        As[i][j] = (__bf16)v;
      }
    }

    // ---------- stage B tile, stored N-major: Bs[n][k] ----------
    if (TRANSB) {
      if (fullN && fullK) {
#pragma unroll
        for (int e = 0; e < 512; e += 128) {
          int idx = e + t;
          int n = idx >> 3;
          int j = (idx & 7) << 2;
          float4 f = *(const float4*)&Bb[(long)(n0 + n) * ldb + (k0 + j)];
          bf16x4 h;
          h[0] = (__bf16)f.x; h[1] = (__bf16)f.y; h[2] = (__bf16)f.z; h[3] = (__bf16)f.w;
          *(bf16x4*)&Bs[n][j] = h;
        }
      } else {
#pragma unroll
        for (int e = 0; e < BN * BK; e += 128) {
          int idx = e + t;
          int n = idx >> 5, j = idx & 31;
          float v = 0.f;
          int gn = n0 + n, gk = k0 + j;
          if (gn < Nc && gk < K) v = Bb[(long)gn * ldb + gk];
          Bs[n][j] = (__bf16)v;
        }
      }
    } else {
      if (fullN && fullK) {
        // pairs along k: coalesced loads along n, packed b32 LDS stores
#pragma unroll
        for (int e = 0; e < 1024; e += 128) {
          int idx = e + t;
          int n = idx & 63;            // fast index -> coalesced global loads
          int j = (idx >> 6) << 1;     // 0,2,...,30
          float x0 = Bb[(long)(k0 + j) * ldb + (n0 + n)];
          float x1 = Bb[(long)(k0 + j + 1) * ldb + (n0 + n)];
          bf16x2 h;
          h[0] = (__bf16)x0; h[1] = (__bf16)x1;
          *(bf16x2*)&Bs[n][j] = h;
        }
      } else {
#pragma unroll
        for (int e = 0; e < BN * BK; e += 128) {
          int idx = e + t;
          int n = idx & 63, j = idx >> 6;
          float v = 0.f;
          int gn = n0 + n, gk = k0 + j;
          if (gn < Nc && gk < K) v = Bb[(long)gk * ldb + gn];
          Bs[n][j] = (__bf16)v;
        }
      }
    }

    // prefetch next A tile into cache (global_prefetch_b8)
    if (k0 + BK < K) {
      int pr = m0 + (t >> 1);
      if (pr < M) __builtin_prefetch(&Ab[(long)pr * lda + k0 + BK], 0, 1);
    }
    __syncthreads();

    // ---- A fragment (16x32 bf16): lane r = row M; K split across lane halves.
    //      elements 0..7  <-> K = half*8 .. half*8+7      (contiguous, 16 B)
    //      elements 8..15 <-> K = 16+half*8 .. 16+half*8+7 (contiguous, 16 B)
    const int arow = (wid << 4) + r;
    v8bf a0 = *(const v8bf*)&As[arow][half << 3];
    v8bf a1 = *(const v8bf*)&As[arow][16 + (half << 3)];
    v16bf af = __builtin_shufflevector(a0, a1, 0, 1, 2, 3, 4, 5, 6, 7,
                                       8, 9, 10, 11, 12, 13, 14, 15);
#pragma unroll
    for (int nt = 0; nt < 4; ++nt) {
      // ---- B fragment (32x16 bf16): lane r = column N; K = half*8..+7, then 16+half*8..+7.
      const int brow = (nt << 4) + r;
      v8bf b0 = *(const v8bf*)&Bs[brow][half << 3];
      v8bf b1 = *(const v8bf*)&Bs[brow][16 + (half << 3)];
      v16bf bf = __builtin_shufflevector(b0, b1, 0, 1, 2, 3, 4, 5, 6, 7,
                                         8, 9, 10, 11, 12, 13, 14, 15);
      acc[nt] = __builtin_amdgcn_wmma_f32_16x16x32_bf16(
          false, af, false, bf, (short)0, acc[nt], false, false);
    }
    __syncthreads();
  }

  // ---- epilogue + store: C/D layout lane = N col, VGPR v = row (+8 for hi half) ----
#pragma unroll
  for (int nt = 0; nt < 4; ++nt) {
    int col = n0 + (nt << 4) + r;
    if (col >= Nc) continue;
    float bv = 0.f;
    if (EPI == EPI_BIAS || EPI == EPI_BIAS_RELU || EPI == EPI_BIAS_SIG) bv = bias[col];
#pragma unroll
    for (int v = 0; v < 8; ++v) {
      int row = m0 + (wid << 4) + (half << 3) + v;
      if (row >= M) continue;
      float x = acc[nt][v];
      if (EPI == EPI_SCALE) x *= scale; else x += bv;
      if (EPI == EPI_BIAS_RELU) x = fmaxf(x, 0.f);
      if (EPI == EPI_BIAS_SIG)  x = 1.f / (1.f + __expf(-x));
      Db[(long)row * ldd + col] = x;
    }
  }
}

// ---------------- helper kernels ----------------
// qkv (D0, D1, 3, H, HD) -> Q/K/V each (D0*H, D1, HD)
__global__ void repack_qkv_k(const float* __restrict__ qkv, float* __restrict__ Q,
                             float* __restrict__ Kd, float* __restrict__ V,
                             int D0, int D1, int H, int HD) {
  long idx = (long)blockIdx.x * blockDim.x + threadIdx.x;
  long total = (long)D0 * D1 * H * HD;
  if (idx >= total) return;
  int d = idx % HD;
  int n = (idx / HD) % D1;
  int h = (idx / ((long)HD * D1)) % H;
  int s = idx / ((long)HD * D1 * H);
  long src = (((long)s * D1 + n) * 3 * H + h) * HD + d;
  Q[idx]  = qkv[src];
  Kd[idx] = qkv[src + (long)H * HD];
  V[idx]  = qkv[src + 2L * H * HD];
}

// X (D0*H, D1, HD) -> Y (D0, D1, H*HD)
__global__ void unpack_heads_k(const float* __restrict__ X, float* __restrict__ Y,
                               int D0, int D1, int H, int HD) {
  long idx = (long)blockIdx.x * blockDim.x + threadIdx.x;
  long total = (long)D0 * D1 * H * HD;
  if (idx >= total) return;
  int d = idx % HD;
  int n = (idx / HD) % D1;
  int h = (idx / ((long)HD * D1)) % H;
  int s = idx / ((long)HD * D1 * H);
  Y[(((long)s * D1 + n) * H + h) * HD + d] = X[idx];
}

// X (A,B,C) -> Y (B,A,C)
__global__ void permute02_k(const float* __restrict__ X, float* __restrict__ Y,
                            int Ad, int Bd, int Cc) {
  long idx = (long)blockIdx.x * blockDim.x + threadIdx.x;
  long total = (long)Ad * Bd * Cc;
  if (idx >= total) return;
  int c = idx % Cc;
  int b = (idx / Cc) % Bd;
  int a = idx / ((long)Cc * Bd);
  Y[((long)b * Ad + a) * Cc + c] = X[idx];
}

// X (N,N,R) -> Y (R,N,N)
__global__ void perm_r_front_k(const float* __restrict__ X, float* __restrict__ Y,
                               int Nn, int Rr) {
  long idx = (long)blockIdx.x * blockDim.x + threadIdx.x;
  long total = (long)Nn * Nn * Rr;
  if (idx >= total) return;
  int rr = idx % Rr;
  int k = (idx / Rr) % Nn;
  int i = idx / ((long)Rr * Nn);
  Y[((long)rr * Nn + i) * Nn + k] = X[idx];
}

// Y(i,j,r) = O(r,i,j) * gate(i,j,r)
__global__ void gate_combine_k(const float* __restrict__ O, const float* __restrict__ gate,
                               float* __restrict__ Y, int Nn, int Rr) {
  long idx = (long)blockIdx.x * blockDim.x + threadIdx.x;
  long total = (long)Nn * Nn * Rr;
  if (idx >= total) return;
  int rr = idx % Rr;
  int j = (idx / Rr) % Nn;
  int i = idx / ((long)Rr * Nn);
  Y[idx] = O[((long)rr * Nn + i) * Nn + j] * gate[idx];
}

__global__ void add2_k(const float* __restrict__ a, const float* __restrict__ b,
                       float* __restrict__ c, int n) {
  int i = blockIdx.x * blockDim.x + threadIdx.x;
  if (i < n) c[i] = a[i] + b[i];
}

// row attention softmax: scores (S*H, N, N), pair bias pb (N,N,H), mask (S,N)
__global__ __launch_bounds__(256) void softmax_row_k(
    float* __restrict__ sc, const float* __restrict__ pb,
    const unsigned char* __restrict__ mask, int H, int Nn) {
  int i = blockIdx.x;
  int b = blockIdx.y;
  int h = b % H, s = b / H;
  long base = ((long)b * Nn + i) * Nn;
  int m = threadIdx.x;
  float v = sc[base + m] + pb[((long)i * Nn + m) * H + h];
  bool ok = mask[(long)s * Nn + i] && mask[(long)s * Nn + m];
  v = ok ? v : -1e9f;
  __shared__ float red[9];
  float mx = block_red_max(v, red);
  float e = __expf(v - mx);
  float sum = block_red_sum(e, red);
  sc[base + m] = e / sum;
}

// col attention softmax: scores (N*H, S, S), mask (S,N)
__global__ __launch_bounds__(64) void softmax_col_k(
    float* __restrict__ sc, const unsigned char* __restrict__ mask,
    int H, int Sn, int Nn) {
  int i = blockIdx.x;
  int b = blockIdx.y;
  int n = b / H;
  long base = ((long)b * Sn + i) * Sn;
  int j = threadIdx.x;
  float v = sc[base + j];
  bool ok = mask[(long)i * Nn + n] && mask[(long)j * Nn + n];
  v = ok ? v : -1e9f;
  __shared__ float red[9];
  float mx = block_red_max(v, red);
  float e = __expf(v - mx);
  float sum = block_red_sum(e, red);
  sc[base + j] = e / sum;
}

// y = LN(x [+ res]) over last dim D; blockDim == D
__global__ void layernorm_k(const float* __restrict__ x, const float* __restrict__ res,
                            const float* __restrict__ g, const float* __restrict__ b,
                            float* __restrict__ y, int D) {
  long row = blockIdx.x;
  int t = threadIdx.x;
  float v = x[row * D + t];
  if (res) v += res[row * D + t];
  __shared__ float red[9];
  float mu = block_red_sum(v, red) / (float)D;
  float d = v - mu;
  float var = block_red_sum(d * d, red) / (float)D;
  y[row * D + t] = d * rsqrtf(var + 1e-5f) * g[t] + b[t];
}

// masked mean over S: msa (S,N,C) -> out (N,C)
__global__ void msa_mean_k(const float* __restrict__ msa, const unsigned char* __restrict__ mask,
                           float* __restrict__ out, int Sn, int Nn, int Cc) {
  int idx = blockIdx.x * blockDim.x + threadIdx.x;
  if (idx >= Nn * Cc) return;
  int n = idx / Cc, c = idx % Cc;
  float acc = 0.f, cnt = 0.f;
  for (int s = 0; s < Sn; ++s) {
    float mk = mask[(long)s * Nn + n] ? 1.f : 0.f;
    acc += msa[(((long)s * Nn + n) * Cc) + c] * mk;
    cnt += mk;
  }
  out[idx] = acc / (cnt + 1e-8f);
}

// pairOut(i,j,:) = pairIn(i,j,:) + LN(lr(i,:)*lr(j,:)); blockDim == P
__global__ __launch_bounds__(128) void opm_add_k(
    const float* __restrict__ lr, const float* __restrict__ g, const float* __restrict__ b,
    const float* __restrict__ pairIn, float* __restrict__ pairOut, int Nn, int Pp) {
  long ij = blockIdx.x;
  int i = (int)(ij / Nn), j = (int)(ij % Nn);
  int p = threadIdx.x;
  float v = lr[(long)i * Pp + p] * lr[(long)j * Pp + p];
  __shared__ float red[9];
  float mu = block_red_sum(v, red) / (float)Pp;
  float d = v - mu;
  float var = block_red_sum(d * d, red) / (float)Pp;
  pairOut[ij * Pp + p] = pairIn[ij * Pp + p] + d * rsqrtf(var + 1e-5f) * g[p] + b[p];
}

// ---------------- host orchestration ----------------
static inline dim3 gemm_grid(int M, int Nc, int batches) {
  return dim3((unsigned)((M + BM - 1) / BM), (unsigned)((Nc + BN - 1) / BN), (unsigned)batches);
}

extern "C" void kernel_launch(void* const* d_in, const int* in_sizes, int n_in,
                              void* d_out, int out_size, void* d_ws, size_t ws_size,
                              hipStream_t stream) {
  (void)in_sizes; (void)n_in; (void)out_size; (void)ws_size;
  const float* msa_in = (const float*)d_in[IN_MSA];
  const float* pair_in = (const float*)d_in[IN_PAIR];
  const unsigned char* mask = (const unsigned char*)d_in[IN_MASK];
  auto PW = [&](int i) { return (const float*)d_in[i]; };

  float* msa_io = (float*)d_out;                 // (S,N,C) working state, final output
  float* pair_io = msa_io + SZ_HEAD;             // (N,N,P) working state, final output

  float* ws = (float*)d_ws;
  float* poolA = ws;                             // 33.55M floats (scores / ptmp / transition h)
  float* poolB = poolA + SZ_SC;                  // 12.58M (qkv / tri buffers / transition out)
  float* poolC = poolB + SZ_QKV;                 // 3 x 4.19M (Q,K,V packed / G)
  float* poolD = poolC + 3 * SZ_HEAD;            // 2 x 4.19M (t0, t1)
  float* poolE = poolD + 2 * SZ_HEAD;            // misc small

  float* qkv = poolB;
  float* Qp = poolC;
  float* Kp = poolC + SZ_HEAD;
  float* Vp = poolC + 2 * SZ_HEAD;
  float* sc = poolA;
  float* t0 = poolD;
  float* t1 = poolD + SZ_HEAD;
  float* pb = poolE;                             // N*N*H = 524288
  float* meanb = pb + (long)NNi * Hn;            // N*C  = 65536
  float* lr = meanb + (long)Ndim * Cdim;         // N*P  = 32768
  float* wsw = lr + (long)Ndim * Pdim;           // P*R  = 4096
  float* wsb = wsw + Pdim * RK;                  // R

  const float scale = 0.17677669529663687f;      // HD^-0.5

  // ================= row attention =================
  gemm_wmma<false, EPI_BIAS><<<gemm_grid(SN, 3 * Cdim, 1), 128, 0, stream>>>(
      msa_in, PW(P_ROW_QKV_W), PW(P_ROW_QKV_B), qkv,
      SN, 3 * Cdim, Cdim, Cdim, 3 * Cdim, 3 * Cdim, 0, 0, 0, 0.f);

  repack_qkv_k<<<(unsigned)((SZ_HEAD + 255) / 256), 256, 0, stream>>>(
      qkv, Qp, Kp, Vp, Sdim, Ndim, Hn, HDm);

  gemm_wmma<false, EPI_BIAS><<<gemm_grid(NNi, Hn, 1), 128, 0, stream>>>(
      pair_in, PW(P_ROW_PB_W), PW(P_ROW_PB_B), pb,
      NNi, Hn, Pdim, Pdim, Hn, Hn, 0, 0, 0, 0.f);

  gemm_wmma<true, EPI_SCALE><<<gemm_grid(Ndim, Ndim, Sdim * Hn), 128, 0, stream>>>(
      Qp, Kp, nullptr, sc,
      Ndim, Ndim, HDm, HDm, HDm, Ndim,
      (long)Ndim * HDm, (long)Ndim * HDm, (long)Ndim * Ndim, scale);

  softmax_row_k<<<dim3(Ndim, Sdim * Hn), 256, 0, stream>>>(sc, pb, mask, Hn, Ndim);

  gemm_wmma<false, EPI_NONE><<<gemm_grid(Ndim, HDm, Sdim * Hn), 128, 0, stream>>>(
      sc, Vp, nullptr, t0,
      Ndim, HDm, Ndim, Ndim, HDm, HDm,
      (long)Ndim * Ndim, (long)Ndim * HDm, (long)Ndim * HDm, 0.f);

  unpack_heads_k<<<(unsigned)((SZ_HEAD + 255) / 256), 256, 0, stream>>>(
      t0, t1, Sdim, Ndim, Hn, HDm);

  gemm_wmma<false, EPI_BIAS><<<gemm_grid(SN, Cdim, 1), 128, 0, stream>>>(
      t1, PW(P_ROW_OUT_W), PW(P_ROW_OUT_B), t0,
      SN, Cdim, Cdim, Cdim, Cdim, Cdim, 0, 0, 0, 0.f);

  layernorm_k<<<SN, Cdim, 0, stream>>>(t0, msa_in, PW(P_ROW_LN_G), PW(P_ROW_LN_B), msa_io, Cdim);

  // ================= col attention =================
  permute02_k<<<(unsigned)((SZ_HEAD + 255) / 256), 256, 0, stream>>>(
      msa_io, t1, Sdim, Ndim, Cdim);                         // (S,N,C)->(N,S,C)

  gemm_wmma<false, EPI_BIAS><<<gemm_grid(SN, 3 * Cdim, 1), 128, 0, stream>>>(
      t1, PW(P_COL_QKV_W), PW(P_COL_QKV_B), qkv,
      SN, 3 * Cdim, Cdim, Cdim, 3 * Cdim, 3 * Cdim, 0, 0, 0, 0.f);

  repack_qkv_k<<<(unsigned)((SZ_HEAD + 255) / 256), 256, 0, stream>>>(
      qkv, Qp, Kp, Vp, Ndim, Sdim, Hn, HDm);                 // -> (N*H, S, HD)

  gemm_wmma<true, EPI_SCALE><<<gemm_grid(Sdim, Sdim, Ndim * Hn), 128, 0, stream>>>(
      Qp, Kp, nullptr, sc,
      Sdim, Sdim, HDm, HDm, HDm, Sdim,
      (long)Sdim * HDm, (long)Sdim * HDm, (long)Sdim * Sdim, scale);

  softmax_col_k<<<dim3(Sdim, Ndim * Hn), 64, 0, stream>>>(sc, mask, Hn, Sdim, Ndim);

  gemm_wmma<false, EPI_NONE><<<gemm_grid(Sdim, HDm, Ndim * Hn), 128, 0, stream>>>(
      sc, Vp, nullptr, t0,
      Sdim, HDm, Sdim, Sdim, HDm, HDm,
      (long)Sdim * Sdim, (long)Sdim * HDm, (long)Sdim * HDm, 0.f);

  unpack_heads_k<<<(unsigned)((SZ_HEAD + 255) / 256), 256, 0, stream>>>(
      t0, t1, Ndim, Sdim, Hn, HDm);                          // (N,S,C)

  gemm_wmma<false, EPI_BIAS><<<gemm_grid(SN, Cdim, 1), 128, 0, stream>>>(
      t1, PW(P_COL_OUT_W), PW(P_COL_OUT_B), t0,
      SN, Cdim, Cdim, Cdim, Cdim, Cdim, 0, 0, 0, 0.f);

  permute02_k<<<(unsigned)((SZ_HEAD + 255) / 256), 256, 0, stream>>>(
      t0, t1, Ndim, Sdim, Cdim);                             // (N,S,C)->(S,N,C)

  layernorm_k<<<SN, Cdim, 0, stream>>>(t1, msa_io, PW(P_COL_LN_G), PW(P_COL_LN_B), msa_io, Cdim);

  // ================= outer product mean =================
  msa_mean_k<<<(Ndim * Cdim + 255) / 256, 256, 0, stream>>>(msa_io, mask, meanb, Sdim, Ndim, Cdim);

  gemm_wmma<false, EPI_BIAS><<<gemm_grid(Ndim, Pdim, 1), 128, 0, stream>>>(
      meanb, PW(P_OPM_W), PW(P_OPM_B), lr,
      Ndim, Pdim, Cdim, Cdim, Pdim, Pdim, 0, 0, 0, 0.f);

  opm_add_k<<<NNi, Pdim, 0, stream>>>(lr, PW(P_OPM_LN_G), PW(P_OPM_LN_B), pair_in, pair_io,
                                      Ndim, Pdim);

  // ================= triangle updates =================
  float* left  = poolB;
  float* right = poolB + SZ_NNR;
  float* gate  = poolB + 2 * SZ_NNR;
  float* Lr    = poolB + 3 * SZ_NNR;
  float* Rr    = poolB + 4 * SZ_NNR;
  float* Obuf  = poolB + 5 * SZ_NNR;
  float* G     = poolC;
  float* ptmp  = poolA;

  auto tri = [&](const float* lw, const float* lb, const float* ew, const float* eb,
                 const float* rw, const float* rb, const float* gw, const float* gb,
                 const float* ow, const float* ob, const float* ng, const float* nb) {
    const float* lw2 = lw;
    const float* lb2 = lb;
    if (ew) {  // left = pr@(l_w+e_w) + (l_b+e_b)
      add2_k<<<(Pdim * RK + 255) / 256, 256, 0, stream>>>(lw, ew, wsw, Pdim * RK);
      add2_k<<<1, 64, 0, stream>>>(lb, eb, wsb, RK);
      lw2 = wsw; lb2 = wsb;
    }
    gemm_wmma<false, EPI_BIAS><<<gemm_grid(NNi, RK, 1), 128, 0, stream>>>(
        pair_io, lw2, lb2, left, NNi, RK, Pdim, Pdim, RK, RK, 0, 0, 0, 0.f);
    gemm_wmma<false, EPI_BIAS><<<gemm_grid(NNi, RK, 1), 128, 0, stream>>>(
        pair_io, rw, rb, right, NNi, RK, Pdim, Pdim, RK, RK, 0, 0, 0, 0.f);
    gemm_wmma<false, EPI_BIAS_SIG><<<gemm_grid(NNi, RK, 1), 128, 0, stream>>>(
        pair_io, gw, gb, gate, NNi, RK, Pdim, Pdim, RK, RK, 0, 0, 0, 0.f);

    perm_r_front_k<<<(unsigned)((SZ_NNR + 255) / 256), 256, 0, stream>>>(left, Lr, Ndim, RK);
    perm_r_front_k<<<(unsigned)((SZ_NNR + 255) / 256), 256, 0, stream>>>(right, Rr, Ndim, RK);

    // out[i,j,r] = sum_k L[i,k,r]*R[k,j,r]: 32 batched 256x256x256 GEMMs
    gemm_wmma<false, EPI_NONE><<<gemm_grid(Ndim, Ndim, RK), 128, 0, stream>>>(
        Lr, Rr, nullptr, Obuf, Ndim, Ndim, Ndim, Ndim, Ndim, Ndim,
        (long)Ndim * Ndim, (long)Ndim * Ndim, (long)Ndim * Ndim, 0.f);

    gate_combine_k<<<(unsigned)((SZ_NNR + 255) / 256), 256, 0, stream>>>(Obuf, gate, G, Ndim, RK);

    gemm_wmma<false, EPI_BIAS><<<gemm_grid(NNi, Pdim, 1), 128, 0, stream>>>(
        G, ow, ob, ptmp, NNi, Pdim, RK, RK, Pdim, Pdim, 0, 0, 0, 0.f);

    layernorm_k<<<NNi, Pdim, 0, stream>>>(ptmp, pair_io, ng, nb, pair_io, Pdim);
  };

  tri(PW(P_TO_L_W), PW(P_TO_L_B), PW(P_TO_E_W), PW(P_TO_E_B),
      PW(P_TO_R_W), PW(P_TO_R_B), PW(P_TO_G_W), PW(P_TO_G_B),
      PW(P_TO_O_W), PW(P_TO_O_B), PW(P_TO_N_G), PW(P_TO_N_B));
  tri(PW(P_TI_L_W), PW(P_TI_L_B), nullptr, nullptr,
      PW(P_TI_R_W), PW(P_TI_R_B), PW(P_TI_G_W), PW(P_TI_G_B),
      PW(P_TI_O_W), PW(P_TI_O_B), PW(P_TI_N_G), PW(P_TI_N_B));

  // ================= transition MLP + final LNs =================
  float* hbuf = poolA;   // NN x 512
  float* out2 = poolB;   // NN x 128
  gemm_wmma<false, EPI_BIAS_RELU><<<gemm_grid(NNi, 4 * Pdim, 1), 128, 0, stream>>>(
      pair_io, PW(P_T1_W), PW(P_T1_B), hbuf,
      NNi, 4 * Pdim, Pdim, Pdim, 4 * Pdim, 4 * Pdim, 0, 0, 0, 0.f);
  gemm_wmma<false, EPI_BIAS><<<gemm_grid(NNi, Pdim, 1), 128, 0, stream>>>(
      hbuf, PW(P_T2_W), PW(P_T2_B), out2,
      NNi, Pdim, 4 * Pdim, 4 * Pdim, Pdim, Pdim, 0, 0, 0, 0.f);
  layernorm_k<<<NNi, Pdim, 0, stream>>>(out2, pair_io, PW(P_TN_G), PW(P_TN_B), pair_io, Pdim);
  layernorm_k<<<NNi, Pdim, 0, stream>>>(pair_io, nullptr, PW(P_PF_G), PW(P_PF_B), pair_io, Pdim);
}
